// MultiHeadSelfAttention_52544629899523
// MI455X (gfx1250) — matrix operand order repaired and checked by reading, not compiled, not run on hardware
//
#include <hip/hip_runtime.h>
#include <hip/hip_bf16.h>

// ---------------- CDNA5 bf16 WMMA building blocks (wave32) ----------------
typedef __attribute__((ext_vector_type(16))) __bf16 v16bf;
typedef __attribute__((ext_vector_type(8)))  __bf16 v8bf;
typedef __attribute__((ext_vector_type(8)))  float  v8f;
typedef __attribute__((ext_vector_type(4)))  unsigned int u32x4;
typedef __attribute__((ext_vector_type(4)))  int i32x4;
typedef __attribute__((ext_vector_type(8)))  int i32x8;

__device__ __forceinline__ v8f wmma_bf16(v16bf a, v16bf b, v8f c) {
  // D = A(16x32 bf16) * B(32x16 bf16) + C(16x16 f32)
  return __builtin_amdgcn_wmma_f32_16x16x32_bf16(false, a, false, b, (short)0, c, false, false);
}

// A fragment: lane = (m = lane&15, g = lane>>4); element e -> K = g*16 + e
__device__ __forceinline__ v16bf load_a_frag(const __bf16* A, int lda, int row0, int k0) {
  int lane = threadIdx.x & 31;
  int m = lane & 15, g = lane >> 4;
  const __bf16* p = A + (size_t)(row0 + m) * lda + k0 + g * 16;
  return *(const v16bf*)p;
}

// B fragment from a row-major M (logical B[k][n] = M[n][k], i.e. y = x * M^T)
__device__ __forceinline__ v16bf load_b_frag(const __bf16* M, int ldm, int col0, int k0) {
  int lane = threadIdx.x & 31;
  int n = lane & 15, g = lane >> 4;
  return *(const v16bf*)(M + (size_t)(col0 + n) * ldm + k0 + g * 16);
}

// B fragment from an LDS-staged [128][32] bf16 tile (row = weight output column).
__device__ __forceinline__ v16bf load_b_frag_lds(const __bf16* Bt, int col0) {
  int lane = threadIdx.x & 31;
  int n = lane & 15, g = lane >> 4;
  return *(const v16bf*)(Bt + (col0 + n) * 32 + g * 16);
}

// B fragment with the source row clamped (causal over-read guard).
__device__ __forceinline__ v16bf load_b_frag_clamprow(const __bf16* M, int ldm, int row0, int k0, int maxrow) {
  int lane = threadIdx.x & 31;
  int r = row0 + (lane & 15);
  if (r > maxrow - 1) r = maxrow - 1;
  int g = lane >> 4;
  return *(const v16bf*)(M + (size_t)r * ldm + k0 + g * 16);
}

// B fragment with the K-offset clamped (row length guard); masked probs zero the garbage.
__device__ __forceinline__ v16bf load_b_frag_clampk(const __bf16* M, int ldm, int row0, int k0, int maxk) {
  int lane = threadIdx.x & 31;
  int n = lane & 15, g = lane >> 4;
  int kk = k0 + g * 16;
  if (kk > maxk - 16) kk = maxk - 16;
  return *(const v16bf*)(M + (size_t)(row0 + n) * ldm + kk);
}

// ---------------- problem constants ----------------
#define BATCH   2
#define SEQ     2048
#define DMODEL  1024
#define NHEADS  16
#define DK      64
#define BS      (BATCH * SEQ)          // 4096 rows
#define LN_THETA_OVER_DK 0.14391157f   // ln(10000)/64

// ---------------- Tensor Data Mover: stage a 128(row) x 32(k) bf16 tile to LDS ----------
// gsrc points at element (row0, k0) of a row-major [1024][1024] bf16 matrix.
// D# per CDNA5 ISA: group0 = {count/lds_addr/global_addr/type},
// group1 = {data_size, tensor dims 1024x1024, tile 32x128, dim0 stride 1024}.
// This toolchain exposes the 6-arg builtin: (g0, g1, g2, g3, g4, cpol).
__device__ __forceinline__ void tdm_load_w_tile(const __bf16* gsrc, void* lds_dst) {
  unsigned long long ga = (unsigned long long)(size_t)gsrc;
  unsigned int la = (unsigned int)(size_t)lds_dst;      // LDS aperture: offset in low 32 bits
  u32x4 g0;
  g0[0] = 1u;                                            // count=1, no gather
  g0[1] = la;                                            // lds_addr [63:32]
  g0[2] = (unsigned int)ga;                              // global_addr [95:64]
  g0[3] = (unsigned int)((ga >> 32) & 0x01FFFFFFu) | (2u << 30);  // addr[56:32] | type=2
  i32x8 g1;
  g1[0] = 0x00010000;                                    // workgroup_mask=0, data_size=1 (2B)
  g1[1] = (int)((DMODEL & 0xFFFF) << 16);                // tensor_dim0[15:0] at bits 63:48
  g1[2] = (int)((DMODEL >> 16) | ((DMODEL & 0xFFFF) << 16)); // dim0 hi | tensor_dim1 lo
  g1[3] = (int)((DMODEL >> 16) | (32 << 16));            // dim1 hi | tile_dim0 = 32 (K elems)
  g1[4] = 128;                                           // tile_dim1 = 128 rows, tile_dim2 = 0
  g1[5] = DMODEL;                                        // tensor_dim0_stride lo32 = 1024
  g1[6] = 0;                                             // stride0 hi | dim1 stride lo (unused, 2D)
  g1[7] = 0;
  i32x4 z4 = {0, 0, 0, 0};                               // groups 2/3 unused (2D tensor)
  i32x8 z8 = {0, 0, 0, 0, 0, 0, 0, 0};                   // extra group (unused, zero)
  __builtin_amdgcn_tensor_load_to_lds(g0, g1, z4, z4, z8, 0);
}

// ---------------- kernel 1: fp32 -> bf16 packing ----------------
__global__ void cvt_bf16_kernel(const float* __restrict__ x,
                                const float* __restrict__ Wq, const float* __restrict__ Wk,
                                const float* __restrict__ Wv, const float* __restrict__ Wo,
                                __bf16* xb, __bf16* wqb, __bf16* wkb, __bf16* wvb, __bf16* wob) {
  const size_t NX = (size_t)BS * DMODEL;
  const size_t NW = (size_t)DMODEL * DMODEL;
  size_t stride = (size_t)gridDim.x * blockDim.x;
  for (size_t i = (size_t)blockIdx.x * blockDim.x + threadIdx.x; i < NX + 4 * NW; i += stride) {
    if (i < NX) {
      xb[i] = (__bf16)x[i];
    } else {
      size_t j = i - NX;
      int w = (int)(j / NW);
      size_t o = j % NW;
      const float* src = (w == 0) ? Wq : (w == 1) ? Wk : (w == 2) ? Wv : Wo;
      __bf16* dst = (w == 0) ? wqb : (w == 1) ? wkb : (w == 2) ? wvb : wob;
      dst[o] = (__bf16)src[o];
    }
  }
}

// ---------------- kernel 2: fused QKV projection + RoPE (TDM-fed, double-buffered) ----
// grid = (BS/128, 3*DMODEL/128); block = 256 (8 waves). Wave = 16 rows x 128 cols.
// Weight macro-tile (128 cols x 32 k) staged to LDS by wave 0 via tensor_load_to_lds.
__global__ __launch_bounds__(256) void qkv_rope_kernel(const __bf16* __restrict__ xb,
                                                       const __bf16* __restrict__ wqb,
                                                       const __bf16* __restrict__ wkb,
                                                       const __bf16* __restrict__ wvb,
                                                       __bf16* __restrict__ Qh,
                                                       __bf16* __restrict__ Kh,
                                                       __bf16* __restrict__ Vt) {
  __shared__ __align__(32) __bf16 Btile[2][128][32];   // 2 x 8 KB double buffer
  int wave = threadIdx.x >> 5;
  int lane = threadIdx.x & 31;
  int nl = lane & 15, g = lane >> 4;
  int row0 = blockIdx.x * 128 + wave * 16;     // row in [0, BS)
  int reg  = blockIdx.y >> 3;                  // 0=Q 1=K 2=V
  int colI = (blockIdx.y & 7) * 128;           // column within [0,1024)
  const __bf16* W = (reg == 0) ? wqb : (reg == 1) ? wkb : wvb;

  if (threadIdx.x < 32)                        // wave 0 drives the TDM pipeline
    tdm_load_w_tile(W + (size_t)colI * DMODEL, &Btile[0][0][0]);

  v8f acc[8] = {};
  for (int ks = 0; ks < DMODEL / 32; ++ks) {
    int k0 = ks * 32;
    if (threadIdx.x < 32) {
      if (k0 + 32 < DMODEL) {
        tdm_load_w_tile(W + (size_t)colI * DMODEL + k0 + 32, &Btile[(ks + 1) & 1][0][0]);
        __builtin_amdgcn_s_wait_tensorcnt(1);  // oldest (current tile) complete
      } else {
        __builtin_amdgcn_s_wait_tensorcnt(0);
      }
    }
    __syncthreads();                           // publish LDS tile to all waves
    v16bf a = load_a_frag(xb, DMODEL, row0, k0);
    const __bf16* Bl = &Btile[ks & 1][0][0];
#pragma unroll
    for (int t = 0; t < 8; ++t) {
      v16bf b = load_b_frag_lds(Bl, t * 16);
      acc[t] = wmma_bf16(a, b, acc[t]);
    }
    __syncthreads();                           // all waves done before buffer reuse
  }

#pragma unroll
  for (int t = 0; t < 8; ++t) {
    int e0 = colI + t * 16;        // feature base, 16-aligned => single head per tile
    int h  = e0 >> 6;
    int dk = (e0 & 63) + nl;       // this lane's feature-within-head
#pragma unroll
    for (int v = 0; v < 8; ++v) {
      int rowg = row0 + v + 8 * g;
      int b = rowg >> 11;          // / SEQ
      int s = rowg & (SEQ - 1);
      float val = acc[t][v];
      if (reg < 2) {               // RoPE on Q and K
        float other = __shfl_xor(val, 1, 32);   // paired even/odd column, same row
        int dkp = dk & ~1;
        float freq = __expf(-(float)dkp * LN_THETA_OVER_DK);
        float ang = (float)s * freq;
        float c, sn;
        __sincosf(ang, &sn, &c);
        val = (dk & 1) ? (other * sn + val * c) : (val * c - other * sn);
      }
      size_t bh = (size_t)b * NHEADS + h;
      if (reg == 0)      Qh[(bh * SEQ + s) * DK + dk] = (__bf16)val;
      else if (reg == 1) Kh[(bh * SEQ + s) * DK + dk] = (__bf16)val;
      else               Vt[(bh * DK + dk) * SEQ + s] = (__bf16)val;   // transposed for P@V B-frags
    }
  }
}

// ---------------- kernel 3: causal flash attention ----------------
// grid = (SEQ/128, BATCH*NHEADS); block = 256 (8 waves). Wave = 16 q-rows, online softmax,
// 32 keys per step: 4 WMMA for Q·K^T (d_k=64) + 4 WMMA for P·V (64 dims).
__global__ __launch_bounds__(256) void attn_kernel(const __bf16* __restrict__ Qh,
                                                   const __bf16* __restrict__ Kh,
                                                   const __bf16* __restrict__ Vt,
                                                   __bf16* __restrict__ AO) {
  __shared__ __align__(32) __bf16 Pshm[8][16][32];   // per-wave C-layout -> A-layout bounce
  int wave = threadIdx.x >> 5;
  int lane = threadIdx.x & 31;
  int nl = lane & 15, g = lane >> 4;
  int bh = blockIdx.y;
  int b = bh >> 4, h = bh & 15;
  const __bf16* Qp = Qh + (size_t)bh * SEQ * DK;
  const __bf16* Kp = Kh + (size_t)bh * SEQ * DK;
  const __bf16* Vp = Vt + (size_t)bh * DK * SEQ;
  int q0 = blockIdx.x * 128 + wave * 16;

  v16bf aq0 = load_a_frag(Qp, DK, q0, 0);    // resident Q fragments (16x64)
  v16bf aq1 = load_a_frag(Qp, DK, q0, 32);

  v8f O[4] = {};
  float mst[8], lst[8];
#pragma unroll
  for (int v = 0; v < 8; ++v) { mst[v] = -1e30f; lst[v] = 0.f; }

  int kmax = q0 + 16;                        // causal: keys < q0+16
  for (int k0 = 0; k0 < kmax; k0 += 32) {
    v8f s0 = {}, s1 = {};
    {
      v16bf bk = load_b_frag_clamprow(Kp, DK, k0, 0, SEQ);
      s0 = wmma_bf16(aq0, bk, s0);
      bk = load_b_frag_clamprow(Kp, DK, k0, 32, SEQ);
      s0 = wmma_bf16(aq1, bk, s0);
      bk = load_b_frag_clamprow(Kp, DK, k0 + 16, 0, SEQ);
      s1 = wmma_bf16(aq0, bk, s1);
      bk = load_b_frag_clamprow(Kp, DK, k0 + 16, 32, SEQ);
      s1 = wmma_bf16(aq1, bk, s1);
    }
#pragma unroll
    for (int v = 0; v < 8; ++v) {
      int q = q0 + v + 8 * g;
      float x0 = s0[v] * 0.125f;             // 1/sqrt(64)
      float x1 = s1[v] * 0.125f;
      if (k0 + nl > q)      x0 = -1e30f;     // causal mask
      if (k0 + 16 + nl > q) x1 = -1e30f;
      float mx = fmaxf(x0, x1);
#pragma unroll
      for (int o = 1; o < 16; o <<= 1) mx = fmaxf(mx, __shfl_xor(mx, o, 32));
      float nm = fmaxf(mst[v], mx);
      float alpha = __expf(mst[v] - nm);
      mst[v] = nm;
      float p0 = __expf(x0 - nm);
      float p1 = __expf(x1 - nm);
      float rs = p0 + p1;
#pragma unroll
      for (int o = 1; o < 16; o <<= 1) rs += __shfl_xor(rs, o, 32);
      lst[v] = lst[v] * alpha + rs;
#pragma unroll
      for (int c = 0; c < 4; ++c) O[c][v] *= alpha;
      Pshm[wave][v + 8 * g][nl]      = (__bf16)p0;   // transpose C-layout -> row-major
      Pshm[wave][v + 8 * g][nl + 16] = (__bf16)p1;
    }
    v16bf ap = load_a_frag(&Pshm[wave][0][0], 32, 0, 0);  // P as A-fragment (same-wave LDS, in-order)
#pragma unroll
    for (int c = 0; c < 4; ++c) {
      v16bf bv = load_b_frag_clampk(Vp, SEQ, c * 16, k0, SEQ);
      O[c] = wmma_bf16(ap, bv, O[c]);
    }
  }

#pragma unroll
  for (int c = 0; c < 4; ++c) {
#pragma unroll
    for (int v = 0; v < 8; ++v) {
      int s = q0 + v + 8 * g;
      float val = O[c][v] / lst[v];
      AO[((size_t)(b * SEQ + s)) * DMODEL + h * DK + c * 16 + nl] = (__bf16)val;
    }
  }
}

// ---------------- kernel 4: output projection (TDM-fed, fp32 out) ----------------
__global__ __launch_bounds__(256) void oproj_kernel(const __bf16* __restrict__ AO,
                                                    const __bf16* __restrict__ wob,
                                                    float* __restrict__ out) {
  __shared__ __align__(32) __bf16 Btile[2][128][32];
  int wave = threadIdx.x >> 5;
  int lane = threadIdx.x & 31;
  int nl = lane & 15, g = lane >> 4;
  int row0 = blockIdx.x * 128 + wave * 16;
  int col0 = blockIdx.y * 128;

  if (threadIdx.x < 32)
    tdm_load_w_tile(wob + (size_t)col0 * DMODEL, &Btile[0][0][0]);

  v8f acc[8] = {};
  for (int ks = 0; ks < DMODEL / 32; ++ks) {
    int k0 = ks * 32;
    if (threadIdx.x < 32) {
      if (k0 + 32 < DMODEL) {
        tdm_load_w_tile(wob + (size_t)col0 * DMODEL + k0 + 32, &Btile[(ks + 1) & 1][0][0]);
        __builtin_amdgcn_s_wait_tensorcnt(1);
      } else {
        __builtin_amdgcn_s_wait_tensorcnt(0);
      }
    }
    __syncthreads();
    v16bf a = load_a_frag(AO, DMODEL, row0, k0);
    const __bf16* Bl = &Btile[ks & 1][0][0];
#pragma unroll
    for (int t = 0; t < 8; ++t) {
      v16bf b = load_b_frag_lds(Bl, t * 16);
      acc[t] = wmma_bf16(a, b, acc[t]);
    }
    __syncthreads();
  }
#pragma unroll
  for (int t = 0; t < 8; ++t)
#pragma unroll
    for (int v = 0; v < 8; ++v)
      out[((size_t)(row0 + v + 8 * g)) * DMODEL + col0 + t * 16 + nl] = acc[t][v];
}

// ---------------- launcher ----------------
extern "C" void kernel_launch(void* const* d_in, const int* in_sizes, int n_in,
                              void* d_out, int out_size, void* d_ws, size_t ws_size,
                              hipStream_t stream) {
  const float* x  = (const float*)d_in[0];
  const float* Wq = (const float*)d_in[1];
  const float* Wk = (const float*)d_in[2];
  const float* Wv = (const float*)d_in[3];
  const float* Wo = (const float*)d_in[4];
  (void)in_sizes; (void)n_in; (void)out_size; (void)ws_size;

  char* ws = (char*)d_ws;
  const size_t MB = 1ull << 20;
  __bf16* xb  = (__bf16*)(ws + 0);        //  8 MB : x bf16 (4096x1024)
  __bf16* wqb = (__bf16*)(ws + 8 * MB);   //  2 MB each weight
  __bf16* wkb = (__bf16*)(ws + 10 * MB);
  __bf16* wvb = (__bf16*)(ws + 12 * MB);
  __bf16* wob = (__bf16*)(ws + 14 * MB);
  __bf16* Qh  = (__bf16*)(ws + 16 * MB);  //  8 MB : [bh][s][dk]
  __bf16* Kh  = (__bf16*)(ws + 24 * MB);  //  8 MB : [bh][s][dk]
  __bf16* Vt  = (__bf16*)(ws + 32 * MB);  //  8 MB : [bh][dk][s]
  __bf16* AO  = (__bf16*)(ws + 40 * MB);  //  8 MB : attention out, row-major (BS x 1024)

  cvt_bf16_kernel<<<4096, 256, 0, stream>>>(x, Wq, Wk, Wv, Wo, xb, wqb, wkb, wvb, wob);

  dim3 g2(BS / 128, 3 * DMODEL / 128);    // (32, 24)
  qkv_rope_kernel<<<g2, 256, 0, stream>>>(xb, wqb, wkb, wvb, Qh, Kh, Vt);

  dim3 g3(SEQ / 128, BATCH * NHEADS);     // (16, 32)
  attn_kernel<<<g3, 256, 0, stream>>>(Qh, Kh, Vt, AO);

  dim3 g4(BS / 128, DMODEL / 128);        // (32, 8)
  oproj_kernel<<<g4, 256, 0, stream>>>(AO, wob, (float*)d_out);
}
